// MultiAttentionHead_26817775796453
// MI455X (gfx1250) — matrix operand order repaired and checked by reading, not compiled, not run on hardware
//
#include <hip/hip_runtime.h>

// Problem constants (from reference)
#define EMBD    768
#define HEADS   12
#define HDIM    64
#define SEQ     2048
#define BATCH   2
// Fold softmax scale AND log2(e) into Q so scores live in the exp2 domain:
// softmax(s/8) == exp2-normalized with s' = s * (0.125 * log2e).
#define QSCALE  (0.125f * 1.44269504088896340736f)
#define QKVELEMS (BATCH * HEADS * SEQ * HDIM)   // 3,145,728
#define WELEMS   (EMBD * EMBD)                  // 589,824

typedef __attribute__((ext_vector_type(16))) __bf16 v16bf;
typedef __attribute__((ext_vector_type(8)))  __bf16 v8bf;
typedef __attribute__((ext_vector_type(8)))  float  v8f;

// fp32 -> bf16 round-to-nearest-even (one-shot paths only)
static __device__ inline __bf16 f2bf(float f) {
  unsigned u = __builtin_bit_cast(unsigned int, f);
  u += 0x7fffu + ((u >> 16) & 1u);
  unsigned short s = (unsigned short)(u >> 16);
  return __builtin_bit_cast(__bf16, s);
}
// fp32 -> bf16 by truncation: 1 shift; store side can fuse to ds_store_b16_d16_hi.
static __device__ inline unsigned short f2bfbits_trunc(float f) {
  return (unsigned short)(__builtin_bit_cast(unsigned int, f) >> 16);
}

// Build a v16bf fragment from two contiguous 16-byte chunks.
static __device__ inline v16bf ld_frag(const __bf16* p0, const __bf16* p1) {
  v8bf lo = *(const v8bf*)p0;
  v8bf hi = *(const v8bf*)p1;
  return __builtin_shufflevector(lo, hi, 0, 1, 2, 3, 4, 5, 6, 7,
                                 8, 9, 10, 11, 12, 13, 14, 15);
}

// Single-instruction float max (avoids canonicalize / cmp+cndmask lowering).
static __device__ inline float vmax(float a, float b) {
  float d;
  asm("v_max_num_f32_e32 %0, %1, %2" : "=v"(d) : "v"(a), "v"(b));
  return d;
}
// Fused DPP16 max steps: x = max(dpp(x), x) in ONE VALU instruction each.
static __device__ inline float dppmax_x1(float x) {  // xor 1
  asm("v_max_num_f32_dpp %0, %0, %0 quad_perm:[1,0,3,2] row_mask:0xf bank_mask:0xf bound_ctrl:1"
      : "+v"(x));
  return x;
}
static __device__ inline float dppmax_x2(float x) {  // xor 2
  asm("v_max_num_f32_dpp %0, %0, %0 quad_perm:[2,3,0,1] row_mask:0xf bank_mask:0xf bound_ctrl:1"
      : "+v"(x));
  return x;
}
static __device__ inline float dppmax_hm(float x) {  // xor 7 (half mirror)
  asm("v_max_num_f32_dpp %0, %0, %0 row_half_mirror row_mask:0xf bank_mask:0xf bound_ctrl:1"
      : "+v"(x));
  return x;
}
static __device__ inline float dppmax_m(float x) {   // xor 15 (mirror)
  asm("v_max_num_f32_dpp %0, %0, %0 row_mirror row_mask:0xf bank_mask:0xf bound_ctrl:1"
      : "+v"(x));
  return x;
}

// ---------------------------------------------------------------------------
// Pre-pass 1: Q*(scale*log2e) -> bf16 [N,D]; K -> bf16 [N,D]; V -> bf16
// transposed [D,N] per (b,h).
// ---------------------------------------------------------------------------
__global__ __launch_bounds__(256) void convert_qkv_kernel(
    const float* __restrict__ q, const float* __restrict__ k,
    const float* __restrict__ v, __bf16* __restrict__ qs,
    __bf16* __restrict__ kb, __bf16* __restrict__ vt) {
  const int idx = blockIdx.x * 256 + threadIdx.x;    // over B*H*SEQ*HDIM
  const int d  = idx & (HDIM - 1);
  const int n  = (idx >> 6) & (SEQ - 1);
  const int bh = idx >> 17;                          // /(SEQ*HDIM)
  qs[idx] = f2bf(q[idx] * QSCALE);
  kb[idx] = f2bf(k[idx]);
  vt[((size_t)bh * HDIM + d) * SEQ + n] = f2bf(v[idx]);
}

// Pre-pass 2: W -> bf16 (row-major; B-fragment reads W rows contiguously).
__global__ __launch_bounds__(256) void convert_w_kernel(
    const float* __restrict__ w, __bf16* __restrict__ wb) {
  const int idx = blockIdx.x * 256 + threadIdx.x;
  wb[idx] = f2bf(w[idx]);
}

// ---------------------------------------------------------------------------
// Kernel 1: flash-attention, 64-key blocks. One wave per 16-row Q tile.
// bf16 operands, f32 WMMA accumulate, exp2-domain online softmax. Row sums
// via WMMA against all-ones B; row max via fused DPP16 max steps.
// Output written bf16 in [B, N, H*D] (fused transpose).
// ---------------------------------------------------------------------------
__global__ __launch_bounds__(32) void attn_fa_kernel(
    const __bf16* __restrict__ qs, const __bf16* __restrict__ kb,
    const __bf16* __restrict__ vt, __bf16* __restrict__ xb) {
  const int lane = threadIdx.x & 31;
  const int hf   = lane >> 4;
  const int ln   = lane & 15;

  const int tile = blockIdx.x;
  const int r0   = (tile & 127) << 4;   // SEQ/16 row tiles
  const int bh   = tile >> 7;
  const int head = bh % HEADS;
  const int b    = bh / HEADS;

  const __bf16* Qp = qs + (size_t)bh * SEQ * HDIM;
  const __bf16* Kp = kb + (size_t)bh * SEQ * HDIM;
  const __bf16* Vp = vt + (size_t)bh * HDIM * SEQ;   // [D, N]
  __bf16*       Xp = xb + (size_t)b * SEQ * EMBD + head * HDIM;

  __shared__ __attribute__((aligned(16))) unsigned short lds_p[16 * 64];

  // Q A-fragments: row M = ln; element e -> K = (e>>3)*16 + hf*8 + (e&7)
  const __bf16* qrow = Qp + (r0 + ln) * HDIM;
  const v16bf a_lo = ld_frag(qrow + hf * 8,      qrow + 16 + hf * 8);
  const v16bf a_hi = ld_frag(qrow + 32 + hf * 8, qrow + 48 + hf * 8);

  // Constant all-ones B fragment for row sums.
  v16bf ones;
#pragma unroll
  for (int e = 0; e < 16; ++e) ones[e] = f2bf(1.0f);

  float rmax[8];
#pragma unroll
  for (int i = 0; i < 8; ++i) rmax[i] = -1e30f;
  v8f o[4] = {};
  v8f osum = {};

  for (int c0 = 0; c0 < SEQ; c0 += 64) {
    if (c0 + 64 < SEQ) {
      __builtin_prefetch(Kp + (c0 + 64 + ln) * HDIM, 0, 3);
      __builtin_prefetch(Vp + ln * SEQ + c0 + 64, 0, 3);
    }

    // S tiles: 4 x (16 rows x 16 keys); B-fragments contiguous along d.
    v8f s[4];
#pragma unroll
    for (int u = 0; u < 4; ++u) {
      const __bf16* kp = Kp + (c0 + u * 16 + ln) * HDIM + hf * 8;
      const v16bf bk_lo = ld_frag(kp,      kp + 16);
      const v16bf bk_hi = ld_frag(kp + 32, kp + 48);
      v8f ss = {};
      ss = __builtin_amdgcn_wmma_f32_16x16x32_bf16(false, a_lo, false, bk_lo, (short)0, ss, false, false);
      s[u] = __builtin_amdgcn_wmma_f32_16x16x32_bf16(false, a_hi, false, bk_hi, (short)0, ss, false, false);
    }

    // Per-row running max: combine 4 tiles, then 4 fused DPP steps per row.
    float m[8];
#pragma unroll
    for (int i = 0; i < 8; ++i)
      m[i] = vmax(vmax(s[0][i], s[1][i]), vmax(s[2][i], s[3][i]));
#pragma unroll
    for (int i = 0; i < 8; ++i) m[i] = dppmax_x1(m[i]);
#pragma unroll
    for (int i = 0; i < 8; ++i) m[i] = dppmax_x2(m[i]);
#pragma unroll
    for (int i = 0; i < 8; ++i) m[i] = dppmax_hm(m[i]);
#pragma unroll
    for (int i = 0; i < 8; ++i) m[i] = dppmax_m(m[i]);

    float corr[8];
#pragma unroll
    for (int i = 0; i < 8; ++i) {
      const float nm = vmax(rmax[i], m[i]);
      corr[i] = __builtin_amdgcn_exp2f(rmax[i] - nm);   // exp2 domain: no mul
      rmax[i] = nm;
      // P = exp2(s' - nm'), staged as truncated bf16 (shift-only conversion).
#pragma unroll
      for (int u = 0; u < 4; ++u)
        lds_p[((hf << 3) + i) * 64 + u * 16 + ln] =
            f2bfbits_trunc(__builtin_amdgcn_exp2f(s[u][i] - nm));
    }
    __syncthreads();

    // P (16x64) in A layout: two fragments, keys [0,32) and [32,64).
    const v8bf* lp = (const v8bf*)lds_p;
    const v16bf ap0 = __builtin_shufflevector(
        lp[ln * 8 + hf], lp[ln * 8 + 2 + hf],
        0, 1, 2, 3, 4, 5, 6, 7, 8, 9, 10, 11, 12, 13, 14, 15);
    const v16bf ap1 = __builtin_shufflevector(
        lp[ln * 8 + 4 + hf], lp[ln * 8 + 6 + hf],
        0, 1, 2, 3, 4, 5, 6, 7, 8, 9, 10, 11, 12, 13, 14, 15);

    // Row sums via WMMA against all-ones B.
#pragma unroll
    for (int i = 0; i < 8; ++i) osum[i] *= corr[i];
    osum = __builtin_amdgcn_wmma_f32_16x16x32_bf16(false, ap0, false, ones, (short)0, osum, false, false);
    osum = __builtin_amdgcn_wmma_f32_16x16x32_bf16(false, ap1, false, ones, (short)0, osum, false, false);

    // O = O*corr + P @ V  (V transposed: contiguous along keys).
#pragma unroll
    for (int t = 0; t < 4; ++t) {
      const __bf16* vp = Vp + (t * 16 + ln) * SEQ + c0 + hf * 8;
      const v16bf bv0 = ld_frag(vp,      vp + 16);
      const v16bf bv1 = ld_frag(vp + 32, vp + 48);
      v8f oo = o[t];
#pragma unroll
      for (int i = 0; i < 8; ++i) oo[i] *= corr[i];
      oo   = __builtin_amdgcn_wmma_f32_16x16x32_bf16(false, ap0, false, bv0, (short)0, oo, false, false);
      o[t] = __builtin_amdgcn_wmma_f32_16x16x32_bf16(false, ap1, false, bv1, (short)0, oo, false, false);
    }
    __syncthreads();
  }

  // Normalize with one reciprocal per row (v_rcp_f32) instead of 32 divides.
  float rinv[8];
#pragma unroll
  for (int i = 0; i < 8; ++i) rinv[i] = __builtin_amdgcn_rcpf(osum[i]);
#pragma unroll
  for (int t = 0; t < 4; ++t)
#pragma unroll
    for (int i = 0; i < 8; ++i) {
      const int row = r0 + (hf << 3) + i;
      Xp[(size_t)row * EMBD + t * 16 + ln] = f2bf(o[t][i] * rinv[i]);
    }
}

// ---------------------------------------------------------------------------
// Kernel 2: projection GEMM  out[B*N, 768] = X @ W^T + bias  (fp32 out).
// One wave per 16x64 strip: A fragment reused across 4 WMMAs per K step.
// ---------------------------------------------------------------------------
__global__ __launch_bounds__(32) void proj_gemm_kernel(
    const __bf16* __restrict__ x, const __bf16* __restrict__ w,
    const float* __restrict__ bias, float* __restrict__ out) {
  const int lane = threadIdx.x & 31;
  const int hf   = lane >> 4;
  const int ln   = lane & 15;
  const int cg   = blockIdx.x % (EMBD / 64);   // 12 column groups of 64
  const int rt   = blockIdx.x / (EMBD / 64);   // 256 row tiles
  const int r0   = rt << 4;
  const int c0   = cg << 6;

  v8f acc[4] = {};
  for (int k0 = 0; k0 < EMBD; k0 += 32) {
    const __bf16* xrow = x + (size_t)(r0 + ln) * EMBD + k0;
    const v16bf a = ld_frag(xrow + hf * 8, xrow + 16 + hf * 8);
#pragma unroll
    for (int u = 0; u < 4; ++u) {
      const __bf16* wrow = w + (size_t)(c0 + u * 16 + ln) * EMBD + k0 + hf * 8;
      const v16bf bw = ld_frag(wrow, wrow + 16);
      acc[u] = __builtin_amdgcn_wmma_f32_16x16x32_bf16(false, a, false, bw, (short)0, acc[u], false, false);
    }
  }
#pragma unroll
  for (int u = 0; u < 4; ++u) {
    const float bb = bias[c0 + u * 16 + ln];
#pragma unroll
    for (int i = 0; i < 8; ++i)
      out[(size_t)(r0 + (hf << 3) + i) * EMBD + c0 + u * 16 + ln] = acc[u][i] + bb;
  }
}

extern "C" void kernel_launch(void* const* d_in, const int* in_sizes, int n_in,
                              void* d_out, int out_size, void* d_ws, size_t ws_size,
                              hipStream_t stream) {
  (void)in_sizes; (void)n_in; (void)out_size; (void)ws_size;
  const float* q  = (const float*)d_in[0];
  const float* k  = (const float*)d_in[1];
  const float* v  = (const float*)d_in[2];
  const float* w  = (const float*)d_in[3];
  const float* b  = (const float*)d_in[4];
  float* out = (float*)d_out;

  // Workspace layout (bf16): Qs | Kb | Vt | Wb | Xb  (~26.3 MB total)
  __bf16* Qs = (__bf16*)d_ws;
  __bf16* Kb = Qs + QKVELEMS;
  __bf16* Vt = Kb + QKVELEMS;
  __bf16* Wb = Vt + QKVELEMS;
  __bf16* Xb = Wb + WELEMS;

  convert_qkv_kernel<<<QKVELEMS / 256, 256, 0, stream>>>(q, k, v, Qs, Kb, Vt);
  convert_w_kernel<<<WELEMS / 256, 256, 0, stream>>>(w, Wb);
  attn_fa_kernel<<<BATCH * HEADS * (SEQ / 16), dim3(32), 0, stream>>>(Qs, Kb, Vt, Xb);
  proj_gemm_kernel<<<(BATCH * SEQ / 16) * (EMBD / 64), dim3(32), 0, stream>>>(Xb, Wb, b, out);
}
